// TemporalRunnerGNN_60894046322803
// MI455X (gfx1250) — compile-verified, matrix operand order checked by reading, not run on hardware
//
#include <hip/hip_runtime.h>

typedef __attribute__((ext_vector_type(16))) _Float16 v16h;
typedef __attribute__((ext_vector_type(8)))  float    v8f;

#define NPER   23
#define EPG    506     // 23*22 edges per frame graph
#define HIDD   128
#define NHEADS 4
#define KRT    8
#define RTD    16
#define TMAX   40
#define TINT   10

// CDNA5 async global->LDS staging (ASYNCcnt-tracked). The load itself is
// inline asm (ISA §15.18: GLOBAL_LOAD_ASYNC_TO_LDS_B128, "vdst, vaddr, off");
// completion is waited with s_wait_asynccnt via the probe-confirmed builtin.
#if defined(__gfx1250__) && __has_builtin(__builtin_amdgcn_s_wait_asynccnt)
#define HAVE_ASYNC_LDS 1
#else
#define HAVE_ASYNC_LDS 0
#endif

#if HAVE_ASYNC_LDS
__device__ __forceinline__ void async_cp16(const float* gptr, const float* ldsptr)
{
    // low 32 bits of a generic shared-pointer == workgroup-relative LDS offset
    unsigned lds_off = (unsigned)(size_t)ldsptr;
    unsigned long long ga = (unsigned long long)(size_t)gptr;
    asm volatile("global_load_async_to_lds_b128 %0, %1, off"
                 :: "v"(lds_off), "v"(ga) : "memory");
}
#endif

// =====================================================================
// Generic WMMA GEMM:  C[M,N] = act(A[M,Kd] @ W[N,Kd]^T + bias[N])
// block = 128 threads = 4 waves; block tile 32(M) x 64(N);
// wave w -> cols [n0+16w, n0+16w+16), two 16x16 WMMA tiles along M.
// LDS staged in f32 (async DMA path), converted to f16 at pack time.
// =====================================================================
__global__ __launch_bounds__(128)
void gemm_wmma_kernel(const float* __restrict__ A, const float* __restrict__ W,
                      const float* __restrict__ bias, float* __restrict__ C,
                      int M, int N, int Kd, int act)
{
    __shared__ float Asf[32 * 32];   // [m][k]  4KB
    __shared__ float Bsf[64 * 32];   // [n][k]  8KB
    const int lane = threadIdx.x & 31;
    const int wave = threadIdx.x >> 5;
    const int m0 = blockIdx.x * 32;
    const int n0 = blockIdx.y * 64;
    v8f acc0 = {}, acc1 = {};
    for (int k0 = 0; k0 < Kd; k0 += 32) {
        if (k0 + 32 < Kd) {   // prefetch next K slab (global_prefetch_b8)
            int pm = m0 + (threadIdx.x & 31); if (pm >= M) pm = M - 1;
            int pn = n0 + (threadIdx.x & 63); if (pn >= N) pn = N - 1;
            __builtin_prefetch(&A[(size_t)pm * Kd + k0 + 32], 0, 1);
            __builtin_prefetch(&W[(size_t)pn * Kd + k0 + 32], 0, 1);
        }
        const bool fullK = (k0 + 32 <= Kd);
#if HAVE_ASYNC_LDS
        if (fullK && (m0 + 32 <= M)) {
            // A tile 32x32 f32 = 4KB = 256 chunks of 16B; 2 chunks/thread
#pragma unroll
            for (int i = 0; i < 2; ++i) {
                int chunk = threadIdx.x + i * 128;
                int row = chunk >> 3, c4 = (chunk & 7) * 4;
                async_cp16(A + (size_t)(m0 + row) * Kd + k0 + c4, &Asf[chunk * 4]);
            }
        } else
#endif
        {
            for (int t = threadIdx.x; t < 32 * 32; t += 128) {
                int m = t >> 5, kk = t & 31;
                float vv = 0.f;
                if ((m0 + m) < M && (k0 + kk) < Kd) vv = A[(size_t)(m0 + m) * Kd + k0 + kk];
                Asf[t] = vv;
            }
        }
#if HAVE_ASYNC_LDS
        if (fullK && (n0 + 64 <= N)) {
            // B tile 64x32 f32 = 8KB = 512 chunks of 16B; 4 chunks/thread
#pragma unroll
            for (int i = 0; i < 4; ++i) {
                int chunk = threadIdx.x + i * 128;
                int row = chunk >> 3, c4 = (chunk & 7) * 4;
                async_cp16(W + (size_t)(n0 + row) * Kd + k0 + c4, &Bsf[chunk * 4]);
            }
        } else
#endif
        {
            for (int t = threadIdx.x; t < 64 * 32; t += 128) {
                int n = t >> 5, kk = t & 31;
                float vv = 0.f;
                if ((n0 + n) < N && (k0 + kk) < Kd) vv = W[(size_t)(n0 + n) * Kd + k0 + kk];
                Bsf[t] = vv;
            }
        }
#if HAVE_ASYNC_LDS
        __builtin_amdgcn_s_wait_asynccnt(0);
#endif
        __syncthreads();
        // Pack fragments per ISA 16-bit layout: lanes 0-15 M/N=0..15 (K grp 0/16),
        // lanes 16-31 same M/N, K offset +8.
        const int mrow = lane & 15;
        const int kb   = (lane >> 4) * 8;
        v16h a0 = {}, a1 = {}, bf = {};
        const float* Ap0 = &Asf[mrow * 32];
        const float* Ap1 = &Asf[(16 + mrow) * 32];
        const float* Bp  = &Bsf[(wave * 16 + mrow) * 32];
#pragma unroll
        for (int r = 0; r < 8; ++r) {
            int kk = ((r & 4) ? 16 : 0) + kb + (r & 3) * 2;
            a0[2 * r] = (_Float16)Ap0[kk];  a0[2 * r + 1] = (_Float16)Ap0[kk + 1];
            a1[2 * r] = (_Float16)Ap1[kk];  a1[2 * r + 1] = (_Float16)Ap1[kk + 1];
            bf[2 * r] = (_Float16)Bp[kk];   bf[2 * r + 1] = (_Float16)Bp[kk + 1];
        }
        acc0 = __builtin_amdgcn_wmma_f32_16x16x32_f16(false, a0, false, bf,
                                                      (short)0, acc0, false, false);
        acc1 = __builtin_amdgcn_wmma_f32_16x16x32_f16(false, a1, false, bf,
                                                      (short)0, acc1, false, false);
        __syncthreads();
    }
    // C layout: VGPR r -> row tile_m0 + r + (lane>=16 ? 8 : 0), col = n-tile + (lane&15)
    int col = n0 + wave * 16 + (lane & 15);
    if (col < N) {
        float bv = bias ? bias[col] : 0.f;
        int rb = ((lane & 16) ? 8 : 0);
#pragma unroll
        for (int r = 0; r < 8; ++r) {
            int row0 = m0 + rb + r;
            if (row0 < M) {
                float o = acc0[r] + bv;
                if (act) o = fmaxf(o, 0.f);
                C[(size_t)row0 * N + col] = o;
            }
            int row1 = m0 + 16 + rb + r;
            if (row1 < M) {
                float o = acc1[r] + bv;
                if (act) o = fmaxf(o, 0.f);
                C[(size_t)row1 * N + col] = o;
            }
        }
    }
}

// =====================================================================
// Unbiased per-column std of edge_attr [E,4]  ->  inv_std[4]
// =====================================================================
__global__ void edge_std_kernel(const float* __restrict__ ea, float* __restrict__ inv_std, int E)
{
    __shared__ float ssum[256], ssq[256];
    int tid = threadIdx.x;
    int col = tid & 3;
    float s = 0.f, q = 0.f;
    for (int r = tid >> 2; r < E; r += 64) {
        float v = ea[(size_t)r * 4 + col];
        s += v; q += v * v;
    }
    ssum[tid] = s; ssq[tid] = q;
    __syncthreads();
    for (int off = 128; off >= 4; off >>= 1) {
        if (tid < off) { ssum[tid] += ssum[tid + off]; ssq[tid] += ssq[tid + off]; }
        __syncthreads();
    }
    if (tid < 4) {
        float mean = ssum[tid] / (float)E;
        float var  = (ssq[tid] - mean * ssum[tid]) / (float)(E - 1);
        float sd   = sqrtf(fmaxf(var, 0.f));
        inv_std[tid] = 1.f / fmaxf(sd, 1e-6f);
    }
}

__global__ void normalize_edges_kernel(const float* __restrict__ ein,
                                       const float* __restrict__ inv_std,
                                       float* __restrict__ eout, int n)
{
    int i = blockIdx.x * blockDim.x + threadIdx.x;
    if (i < n) eout[i] = ein[i] * inv_std[i & 3];
}

// =====================================================================
// TransformerConv edge attention: one wave per dst node, loops 4 heads,
// 22 in-neighbors, e = ea @ e_w^T recomputed on the fly (4 MAC/elt).
// out = relu(mean_heads(msg) + skip)
// =====================================================================
__global__ void tconv_attn_kernel(const float* __restrict__ q, const float* __restrict__ k,
                                  const float* __restrict__ v, const float* __restrict__ ea,
                                  const float* __restrict__ e_w, const float* __restrict__ skip,
                                  float* __restrict__ out, int Nn)
{
    int node = blockIdx.x * (blockDim.x >> 5) + (threadIdx.x >> 5);
    if (node >= Nn) return;
    int lane = threadIdx.x & 31;
    int g  = node / NPER;
    int il = node - g * NPER;
    int ebase = g * EPG + il * 22;
    int ch = lane * 4;
    float accmean[4] = {0.f, 0.f, 0.f, 0.f};
    for (int h = 0; h < NHEADS; ++h) {
        float4 ew0 = *(const float4*)(e_w + (size_t)(h * HIDD + ch + 0) * 4);
        float4 ew1 = *(const float4*)(e_w + (size_t)(h * HIDD + ch + 1) * 4);
        float4 ew2 = *(const float4*)(e_w + (size_t)(h * HIDD + ch + 2) * 4);
        float4 ew3 = *(const float4*)(e_w + (size_t)(h * HIDD + ch + 3) * 4);
        float4 qi  = *(const float4*)(q + (size_t)node * 512 + h * HIDD + ch);
        float alpha[22];
        float amax = -3.0e38f;
#pragma unroll
        for (int jn = 0; jn < 22; ++jn) {
            int jl = jn < il ? jn : jn + 1;
            int jnode = g * NPER + jl;
            float4 ed = *(const float4*)(ea + (size_t)(ebase + jn) * 4);
            float4 kv = *(const float4*)(k + (size_t)jnode * 512 + h * HIDD + ch);
            float e0 = ew0.x * ed.x + ew0.y * ed.y + ew0.z * ed.z + ew0.w * ed.w;
            float e1 = ew1.x * ed.x + ew1.y * ed.y + ew1.z * ed.z + ew1.w * ed.w;
            float e2 = ew2.x * ed.x + ew2.y * ed.y + ew2.z * ed.z + ew2.w * ed.w;
            float e3 = ew3.x * ed.x + ew3.y * ed.y + ew3.z * ed.z + ew3.w * ed.w;
            float part = qi.x * (kv.x + e0) + qi.y * (kv.y + e1)
                       + qi.z * (kv.z + e2) + qi.w * (kv.w + e3);
            for (int off = 16; off > 0; off >>= 1) part += __shfl_xor(part, off, 32);
            float a = part * 0.08838834764831843f;   // 1/sqrt(128)
            alpha[jn] = a;
            amax = fmaxf(amax, a);
        }
        float den = 0.f;
#pragma unroll
        for (int jn = 0; jn < 22; ++jn) { alpha[jn] = __expf(alpha[jn] - amax); den += alpha[jn]; }
        float inv = 1.f / den;
        float a0 = 0.f, a1 = 0.f, a2 = 0.f, a3 = 0.f;
#pragma unroll
        for (int jn = 0; jn < 22; ++jn) {
            int jl = jn < il ? jn : jn + 1;
            int jnode = g * NPER + jl;
            float w = alpha[jn] * inv;
            float4 ed = *(const float4*)(ea + (size_t)(ebase + jn) * 4);
            float4 vv = *(const float4*)(v + (size_t)jnode * 512 + h * HIDD + ch);
            float e0 = ew0.x * ed.x + ew0.y * ed.y + ew0.z * ed.z + ew0.w * ed.w;
            float e1 = ew1.x * ed.x + ew1.y * ed.y + ew1.z * ed.z + ew1.w * ed.w;
            float e2 = ew2.x * ed.x + ew2.y * ed.y + ew2.z * ed.z + ew2.w * ed.w;
            float e3 = ew3.x * ed.x + ew3.y * ed.y + ew3.z * ed.z + ew3.w * ed.w;
            a0 += w * (vv.x + e0); a1 += w * (vv.y + e1);
            a2 += w * (vv.z + e2); a3 += w * (vv.w + e3);
        }
        accmean[0] += 0.25f * a0; accmean[1] += 0.25f * a1;
        accmean[2] += 0.25f * a2; accmean[3] += 0.25f * a3;
    }
    float4 sk = *(const float4*)(skip + (size_t)node * HIDD + ch);
    float4 o;
    o.x = fmaxf(accmean[0] + sk.x, 0.f);
    o.y = fmaxf(accmean[1] + sk.y, 0.f);
    o.z = fmaxf(accmean[2] + sk.z, 0.f);
    o.w = fmaxf(accmean[3] + sk.w, 0.f);
    *(float4*)(out + (size_t)node * HIDD + ch) = o;
}

// =====================================================================
// Sequence build / bias / PE helpers
// =====================================================================
__global__ void build_seq_kernel(const float* __restrict__ h, const int* __restrict__ runner,
                                 const int* __restrict__ lengths, float* __restrict__ seq, int B)
{
    int idx = blockIdx.x * blockDim.x + threadIdx.x;
    int total = B * TMAX * HIDD;
    if (idx >= total) return;
    int c = idx & (HIDD - 1);
    int t = (idx / HIDD) % TMAX;
    int b = idx / (HIDD * TMAX);
    int start = 0;
    for (int i = 0; i < b; ++i) start += lengths[i];
    int len = lengths[b];
    float val = 0.f;
    if (t < len) {
        int gidx = start + t;
        int node = gidx * NPER + runner[gidx];
        val = h[(size_t)node * HIDD + c];
    }
    seq[idx] = val;
}

__global__ void build_bias_kernel(const int* __restrict__ lengths, float* __restrict__ bias, int B)
{
    int idx = blockIdx.x * blockDim.x + threadIdx.x;
    int total = B * TMAX * TMAX;
    if (idx >= total) return;
    int s = idx % TMAX;
    int t = (idx / TMAX) % TMAX;
    int b = idx / (TMAX * TMAX);
    bias[idx] = (s > t || s >= lengths[b]) ? -1e9f : 0.f;
}

// xo[b*T + t, :] = seq[b, t, :] + pe[t, :]   (T <= 40)
__global__ void seq_pe_kernel(const float* __restrict__ seq, const float* __restrict__ pe,
                              float* __restrict__ xo, int B, int T)
{
    int idx = blockIdx.x * blockDim.x + threadIdx.x;
    int total = B * T * HIDD;
    if (idx >= total) return;
    int c = idx & (HIDD - 1);
    int t = (idx / HIDD) % T;
    int b = idx / (HIDD * T);
    xo[idx] = seq[((size_t)b * TMAX + t) * HIDD + c] + pe[(size_t)t * HIDD + c];
}

// xp[(k*B+b)*T + t, 0:128]=seq, [128:144]=emb[k]
__global__ void build_xp_kernel(const float* __restrict__ seq, const float* __restrict__ emb,
                                float* __restrict__ xp, int B)
{
    int idx = blockIdx.x * blockDim.x + threadIdx.x;
    int total = KRT * B * TMAX * (HIDD + RTD);
    if (idx >= total) return;
    int c  = idx % (HIDD + RTD);
    int t  = (idx / (HIDD + RTD)) % TMAX;
    int kb = idx / ((HIDD + RTD) * TMAX);
    int kk = kb / B, b = kb % B;
    xp[idx] = (c < HIDD) ? seq[((size_t)b * TMAX + t) * HIDD + c]
                         : emb[(size_t)kk * RTD + (c - HIDD)];
}

__global__ void add_pe_kernel(float* __restrict__ x, const float* __restrict__ pe, int rows)
{
    int idx = blockIdx.x * blockDim.x + threadIdx.x;
    int total = rows * HIDD;
    if (idx >= total) return;
    int c = idx & (HIDD - 1);
    int t = (idx / HIDD) % TMAX;
    x[idx] += pe[(size_t)t * HIDD + c];
}

// =====================================================================
// Small MHA (dh=32, 4 heads) — one thread per (b,h,t) row, templated T.
// =====================================================================
template <int T>
__global__ void mha_kernel(const float* __restrict__ qkv, const float* __restrict__ bias,
                           float* __restrict__ o, int Bq, int biasRow, int biasBat, int biasMod)
{
    int idx = blockIdx.x * blockDim.x + threadIdx.x;
    if (idx >= Bq * NHEADS * T) return;
    int t = idx % T;
    int h = (idx / T) & 3;
    int b = idx / (NHEADS * T);
    const float* qrow = qkv + ((size_t)b * T + t) * 384 + h * 32;
    const float* brow = bias + (size_t)(b % biasMod) * biasBat + (size_t)t * biasRow;
    float sc[T];
    float mx = -3.0e38f;
#pragma unroll
    for (int s = 0; s < T; ++s) {
        const float* krow = qkv + ((size_t)b * T + s) * 384 + 128 + h * 32;
        float d = 0.f;
#pragma unroll
        for (int c = 0; c < 32; ++c) d += qrow[c] * krow[c];
        d = d * 0.17677669529663687f + brow[s];   // 1/sqrt(32)
        sc[s] = d;
        mx = fmaxf(mx, d);
    }
    float den = 0.f;
#pragma unroll
    for (int s = 0; s < T; ++s) { sc[s] = __expf(sc[s] - mx); den += sc[s]; }
    float inv = 1.f / den;
    float acc[32];
#pragma unroll
    for (int c = 0; c < 32; ++c) acc[c] = 0.f;
#pragma unroll
    for (int s = 0; s < T; ++s) {
        float w = sc[s] * inv;
        const float* vrow = qkv + ((size_t)b * T + s) * 384 + 256 + h * 32;
#pragma unroll
        for (int c = 0; c < 32; ++c) acc[c] += w * vrow[c];
    }
    float* orow = o + ((size_t)b * T + t) * 128 + h * 32;
#pragma unroll
    for (int c = 0; c < 32; ++c) orow[c] = acc[c];
}

// x = LN(x + r) * g + b     one wave per 128-wide row
__global__ void add_ln_kernel(float* __restrict__ x, const float* __restrict__ r,
                              const float* __restrict__ g, const float* __restrict__ b, int M)
{
    int row = blockIdx.x * (blockDim.x >> 5) + (threadIdx.x >> 5);
    if (row >= M) return;
    int lane = threadIdx.x & 31;
    float4 xv = *(const float4*)(x + (size_t)row * HIDD + lane * 4);
    float4 rv = *(const float4*)(r + (size_t)row * HIDD + lane * 4);
    float v0 = xv.x + rv.x, v1 = xv.y + rv.y, v2 = xv.z + rv.z, v3 = xv.w + rv.w;
    float s = v0 + v1 + v2 + v3;
    for (int off = 16; off > 0; off >>= 1) s += __shfl_xor(s, off, 32);
    float mu = s * (1.f / 128.f);
    v0 -= mu; v1 -= mu; v2 -= mu; v3 -= mu;
    float q = v0 * v0 + v1 * v1 + v2 * v2 + v3 * v3;
    for (int off = 16; off > 0; off >>= 1) q += __shfl_xor(q, off, 32);
    float is = rsqrtf(q * (1.f / 128.f) + 1e-5f);
    float4 gv = *(const float4*)(g + lane * 4);
    float4 bv = *(const float4*)(b + lane * 4);
    float4 o;
    o.x = v0 * is * gv.x + bv.x; o.y = v1 * is * gv.y + bv.y;
    o.z = v2 * is * gv.z + bv.z; o.w = v3 * is * gv.w + bv.w;
    *(float4*)(x + (size_t)row * HIDD + lane * 4) = o;
}

// logits/probs head: thread b computes K logits + softmax from row (b*rowStride+rowOff)
__global__ void head_softmax_kernel(const float* __restrict__ xin, int rowStride, int rowOff,
                                    const float* __restrict__ W, const float* __restrict__ bb,
                                    float* __restrict__ logits, float* __restrict__ probs, int B)
{
    int b = threadIdx.x;
    if (b >= B) return;
    const float* xr = xin + ((size_t)b * rowStride + rowOff) * HIDD;
    float lg[KRT];
    float mx = -3.0e38f;
#pragma unroll
    for (int kk = 0; kk < KRT; ++kk) {
        float d = bb[kk];
        for (int c = 0; c < HIDD; ++c) d += xr[c] * W[(size_t)kk * HIDD + c];
        lg[kk] = d;
        mx = fmaxf(mx, d);
    }
    float den = 0.f;
    float ex[KRT];
#pragma unroll
    for (int kk = 0; kk < KRT; ++kk) { ex[kk] = __expf(lg[kk] - mx); den += ex[kk]; }
    float inv = 1.f / den;
#pragma unroll
    for (int kk = 0; kk < KRT; ++kk) {
        logits[(size_t)b * KRT + kk] = lg[kk];
        probs[(size_t)b * KRT + kk]  = ex[kk] * inv;
    }
}

// paths[b,k,t,xy] = path_out[(k*B+b)*T + t] . pos_w[xy] + pos_b[xy]
__global__ void paths_kernel(const float* __restrict__ xin, const float* __restrict__ W,
                             const float* __restrict__ bb, float* __restrict__ outp, int B)
{
    int idx = blockIdx.x * blockDim.x + threadIdx.x;
    int total = B * KRT * TMAX * 2;
    if (idx >= total) return;
    int xy = idx & 1;
    int t  = (idx >> 1) % TMAX;
    int kk = ((idx >> 1) / TMAX) % KRT;
    int b  = idx / (2 * TMAX * KRT);
    const float* xr = xin + (((size_t)kk * B + b) * TMAX + t) * HIDD;
    float d = bb[xy];
    for (int c = 0; c < HIDD; ++c) d += xr[c] * W[(size_t)xy * HIDD + c];
    outp[(((size_t)b * KRT + kk) * TMAX + t) * 2 + xy] = d;
}

// =====================================================================
// Host side
// =====================================================================
struct EncLayer {
    const float *in_w, *in_b, *out_w, *out_b, *ff1_w, *ff1_b, *ff2_w, *ff2_b;
    const float *ln1_g, *ln1_b, *ln2_g, *ln2_b;
};

static inline void launch_gemm(const float* A, const float* W, const float* bias, float* C,
                               int M, int N, int Kd, int act, hipStream_t s)
{
    dim3 grid((M + 31) / 32, (N + 63) / 64);
    gemm_wmma_kernel<<<grid, 128, 0, s>>>(A, W, bias, C, M, N, Kd, act);
}

static void run_encoder(float* xbuf, int Bq, int T, const float* bias,
                        int biasRow, int biasBat, int biasMod, const EncLayer* L,
                        float* qkv, float* obuf, float* o2, float* tmp, hipStream_t s)
{
    int M = Bq * T;
    for (int l = 0; l < 2; ++l) {
        launch_gemm(xbuf, L[l].in_w, L[l].in_b, qkv, M, 384, HIDD, 0, s);
        int th = Bq * NHEADS * T;
        if (T == TINT)
            mha_kernel<TINT><<<(th + 63) / 64, 64, 0, s>>>(qkv, bias, obuf, Bq, biasRow, biasBat, biasMod);
        else
            mha_kernel<TMAX><<<(th + 63) / 64, 64, 0, s>>>(qkv, bias, obuf, Bq, biasRow, biasBat, biasMod);
        launch_gemm(obuf, L[l].out_w, L[l].out_b, o2, M, HIDD, HIDD, 0, s);
        add_ln_kernel<<<(M + 7) / 8, 256, 0, s>>>(xbuf, o2, L[l].ln1_g, L[l].ln1_b, M);
        launch_gemm(xbuf, L[l].ff1_w, L[l].ff1_b, tmp, M, 512, HIDD, 1, s);
        launch_gemm(tmp, L[l].ff2_w, L[l].ff2_b, o2, M, HIDD, 512, 0, s);
        add_ln_kernel<<<(M + 7) / 8, 256, 0, s>>>(xbuf, o2, L[l].ln2_g, L[l].ln2_b, M);
    }
}

extern "C" void kernel_launch(void* const* d_in, const int* in_sizes, int n_in,
                              void* d_out, int out_size, void* d_ws, size_t ws_size,
                              hipStream_t stream)
{
    // ---- inputs (insertion-order pytree flatten of setup_inputs()) ----
    const float* x         = (const float*)d_in[0];   // [N,12]
    const float* edge_attr = (const float*)d_in[1];   // [E,4]
    const float* g1_q_w = (const float*)d_in[2];  const float* g1_q_b = (const float*)d_in[3];
    const float* g1_k_w = (const float*)d_in[4];  const float* g1_k_b = (const float*)d_in[5];
    const float* g1_v_w = (const float*)d_in[6];  const float* g1_v_b = (const float*)d_in[7];
    const float* g1_e_w = (const float*)d_in[8];
    const float* g1_s_w = (const float*)d_in[9];  const float* g1_s_b = (const float*)d_in[10];
    const float* g2_q_w = (const float*)d_in[11]; const float* g2_q_b = (const float*)d_in[12];
    const float* g2_k_w = (const float*)d_in[13]; const float* g2_k_b = (const float*)d_in[14];
    const float* g2_v_w = (const float*)d_in[15]; const float* g2_v_b = (const float*)d_in[16];
    const float* g2_e_w = (const float*)d_in[17];
    const float* g2_s_w = (const float*)d_in[18]; const float* g2_s_b = (const float*)d_in[19];
    EncLayer L[2];
    for (int l = 0; l < 2; ++l) {
        int base = 20 + 12 * l;
        L[l].in_w  = (const float*)d_in[base + 0];  L[l].in_b  = (const float*)d_in[base + 1];
        L[l].out_w = (const float*)d_in[base + 2];  L[l].out_b = (const float*)d_in[base + 3];
        L[l].ff1_w = (const float*)d_in[base + 4];  L[l].ff1_b = (const float*)d_in[base + 5];
        L[l].ff2_w = (const float*)d_in[base + 6];  L[l].ff2_b = (const float*)d_in[base + 7];
        L[l].ln1_g = (const float*)d_in[base + 8];  L[l].ln1_b = (const float*)d_in[base + 9];
        L[l].ln2_g = (const float*)d_in[base + 10]; L[l].ln2_b = (const float*)d_in[base + 11];
    }
    const float* pe     = (const float*)d_in[44];
    const float* proj_w = (const float*)d_in[45]; const float* proj_b = (const float*)d_in[46];
    const float* rt_w   = (const float*)d_in[47]; const float* rt_b   = (const float*)d_in[48];
    const float* emb    = (const float*)d_in[49];
    const float* pos_w  = (const float*)d_in[50]; const float* pos_b  = (const float*)d_in[51];
    const float* shot_w = (const float*)d_in[52]; const float* shot_b = (const float*)d_in[53];
    const int* runner_idx = (const int*)d_in[56];
    const int* lengths    = (const int*)d_in[57];
    (void)n_in; (void)ws_size;

    const int Nn = in_sizes[0] / 12;        // 6164 nodes
    const int E  = in_sizes[1] / 4;         // 135608 edges
    const int B  = in_sizes[57];            // 8 runs
    float* out = (float*)d_out;
    (void)out_size;

    // ---- workspace layout (floats) ----
    float* w0      = (float*)d_ws;
    float* inv_std = w0;                               // 16
    float* ea      = inv_std + 16;                     // E*4
    float* skipb   = ea + (size_t)E * 4;               // Nn*128
    float* hbuf    = skipb + (size_t)Nn * HIDD;        // Nn*128
    float* seq     = hbuf + (size_t)Nn * HIDD;         // B*40*128
    float* biasb   = seq + (size_t)B * TMAX * HIDD;    // B*40*40
    float* region  = biasb + (size_t)B * TMAX * TMAX;  // max(3*Nn*512, encoder scratch)
    float* qb = region;
    float* kb = qb + (size_t)Nn * 512;
    float* vb = kb + (size_t)Nn * 512;
    const int MAXM = KRT * B * TMAX;                   // 2560
    float* enc_x   = region;                           // aliases qkv region (free after GNN)
    float* enc_qkv = enc_x + (size_t)MAXM * HIDD;
    float* enc_o   = enc_qkv + (size_t)MAXM * 384;
    float* enc_o2  = enc_o + (size_t)MAXM * HIDD;
    float* enc_tmp = enc_o2 + (size_t)MAXM * HIDD;
    float* xp      = enc_tmp + (size_t)MAXM * 512;

    // ---- 1) edge-attr normalization ----
    edge_std_kernel<<<1, 256, 0, stream>>>(edge_attr, inv_std, E);
    normalize_edges_kernel<<<(E * 4 + 255) / 256, 256, 0, stream>>>(edge_attr, inv_std, ea, E * 4);

    // ---- 2) GNN layer 1 (in=12) ----
    launch_gemm(x, g1_q_w, g1_q_b, qb, Nn, 512, 12, 0, stream);
    launch_gemm(x, g1_k_w, g1_k_b, kb, Nn, 512, 12, 0, stream);
    launch_gemm(x, g1_v_w, g1_v_b, vb, Nn, 512, 12, 0, stream);
    launch_gemm(x, g1_s_w, g1_s_b, skipb, Nn, HIDD, 12, 0, stream);
    tconv_attn_kernel<<<(Nn + 7) / 8, 256, 0, stream>>>(qb, kb, vb, ea, g1_e_w, skipb, hbuf, Nn);

    // ---- 3) GNN layer 2 (in=128) ----
    launch_gemm(hbuf, g2_q_w, g2_q_b, qb, Nn, 512, HIDD, 0, stream);
    launch_gemm(hbuf, g2_k_w, g2_k_b, kb, Nn, 512, HIDD, 0, stream);
    launch_gemm(hbuf, g2_v_w, g2_v_b, vb, Nn, 512, HIDD, 0, stream);
    launch_gemm(hbuf, g2_s_w, g2_s_b, skipb, Nn, HIDD, HIDD, 0, stream);
    tconv_attn_kernel<<<(Nn + 7) / 8, 256, 0, stream>>>(qb, kb, vb, ea, g2_e_w, skipb, hbuf, Nn);

    // ---- 4) runner sequences + attention bias ----
    build_seq_kernel<<<(B * TMAX * HIDD + 255) / 256, 256, 0, stream>>>(hbuf, runner_idx, lengths, seq, B);
    build_bias_kernel<<<(B * TMAX * TMAX + 255) / 256, 256, 0, stream>>>(lengths, biasb, B);

    // ---- 5) intent head (T=10) ----
    seq_pe_kernel<<<(B * TINT * HIDD + 255) / 256, 256, 0, stream>>>(seq, pe, enc_x, B, TINT);
    run_encoder(enc_x, B, TINT, biasb, TMAX, TMAX * TMAX, B, L, enc_qkv, enc_o, enc_o2, enc_tmp, stream);
    head_softmax_kernel<<<1, 32, 0, stream>>>(enc_x, TINT, TINT - 1, rt_w, rt_b,
                                              out, out + (size_t)B * KRT, B);

    // ---- 6) shot head (T=40) ----
    seq_pe_kernel<<<(B * TMAX * HIDD + 255) / 256, 256, 0, stream>>>(seq, pe, enc_x, B, TMAX);
    run_encoder(enc_x, B, TMAX, biasb, TMAX, TMAX * TMAX, B, L, enc_qkv, enc_o, enc_o2, enc_tmp, stream);
    head_softmax_kernel<<<1, 32, 0, stream>>>(enc_x, TMAX, TMAX - 1, shot_w, shot_b,
                                              out + (size_t)2 * B * KRT, out + (size_t)3 * B * KRT, B);

    // ---- 7) path decode (K folded into batch: Bq = K*B = 64) ----
    build_xp_kernel<<<(KRT * B * TMAX * (HIDD + RTD) + 255) / 256, 256, 0, stream>>>(seq, emb, xp, B);
    launch_gemm(xp, proj_w, proj_b, enc_x, KRT * B * TMAX, HIDD, HIDD + RTD, 0, stream);
    add_pe_kernel<<<(KRT * B * TMAX * HIDD + 255) / 256, 256, 0, stream>>>(enc_x, pe, KRT * B * TMAX);
    run_encoder(enc_x, KRT * B, TMAX, biasb, TMAX, TMAX * TMAX, B, L, enc_qkv, enc_o, enc_o2, enc_tmp, stream);
    paths_kernel<<<(B * KRT * TMAX * 2 + 255) / 256, 256, 0, stream>>>(enc_x, pos_w, pos_b,
                                                                       out + (size_t)4 * B * KRT, B);
}
